// APPNP2_14491219657220
// MI455X (gfx1250) — matrix operand order repaired and checked by reading, not compiled, not run on hardware
//
#include <hip/hip_runtime.h>
#include <cstdint>
#include <cstddef>

typedef __attribute__((ext_vector_type(16))) _Float16 v16h;
typedef __attribute__((ext_vector_type(8)))  _Float16 h8;
typedef __attribute__((ext_vector_type(8)))  float    v8f;
typedef __attribute__((ext_vector_type(4)))  float    f4;
typedef __attribute__((ext_vector_type(2)))  float    f2;

#define N_IN   512
#define N_HID  256
#define N_OUT  64
#define XS_LD  520          // 512 + 8 f16 pad: bank stride 4 dwords, 16B aligned
#define H1_LD  264          // 256 + 8 f16 pad
#define ALPHA  0.1f

// ---------------------------------------------------------------------------
// Fused 2-layer MLP: per block, 16 nodes.
//   Stage 0: 16 waves cooperatively stage x[16x512] -> LDS as f16 (one cvt).
//   Stage 1: wave w computes h1[16, 16w..16w+15] = relu(x@W1^T+b1) via
//            v_wmma_f32_16x16x32_f16 (A from LDS, B = W1 rows from global).
//   Stage 2: waves 0..3 compute h[16, 16w..16w+15] = h1@W2^T+b2 from LDS.
// Writes h (workspace) and z0 = h (d_out).
// ---------------------------------------------------------------------------
__global__ __launch_bounds__(512) void k_mlp(
    const float* __restrict__ x, const _Float16* __restrict__ w1h,
    const float* __restrict__ b1, const _Float16* __restrict__ w2h,
    const float* __restrict__ b2, float* __restrict__ h, float* __restrict__ z)
{
  __shared__ _Float16 xs [16 * XS_LD];   // f16 x tile
  __shared__ _Float16 h1s[16 * H1_LD];   // f16 hidden tile
  const int lane = threadIdx.x & 31;
  const int wave = threadIdx.x >> 5;   // 0..15
  const int half = lane >> 4;          // 0/1
  const int mr   = lane & 15;          // A: row M, B/C: col N
  const size_t m0 = (size_t)blockIdx.x * 16;

  // ---------------- stage 0: x tile -> LDS (f32 -> f16 once) ----------------
  {
    const int row = wave;              // one row per wave
    const int col = lane * 16;         // 32 lanes x 16 elems = 512
    const float* sp = x + (m0 + row) * N_IN + col;
    f4 v0 = *(const f4*)(sp);
    f4 v1 = *(const f4*)(sp + 4);
    f4 v2 = *(const f4*)(sp + 8);
    f4 v3 = *(const f4*)(sp + 12);
    h8 lo, hi;
    #pragma unroll
    for (int e = 0; e < 4; ++e) {
      lo[e]     = (_Float16)v0[e];
      lo[4 + e] = (_Float16)v1[e];
      hi[e]     = (_Float16)v2[e];
      hi[4 + e] = (_Float16)v3[e];
    }
    *(h8*)(&xs[row * XS_LD + col])     = lo;
    *(h8*)(&xs[row * XS_LD + col + 8]) = hi;
  }
  __syncthreads();

  // ---------------- stage 1 ----------------
  const _Float16* w1row = w1h + (size_t)(wave * 16 + mr) * N_IN;
  v8f c = {};
  #pragma unroll
  for (int k0 = 0; k0 < N_IN; k0 += 32) {
    v16h a, b;
    // A fragment from LDS: K = k0 + (e/8)*16 + half*8 + e%8
    h8 al = *(const h8*)(&xs[mr * XS_LD + k0 + half * 8]);
    h8 ah = *(const h8*)(&xs[mr * XS_LD + k0 + 16 + half * 8]);
    // B fragment: K = k0 + (e/8)*16 + half*8 + e%8 (W1 row n = wave*16+mr)
    h8 bl = *(const h8*)(w1row + k0 + half * 8);
    h8 bh = *(const h8*)(w1row + k0 + 16 + half * 8);
    #pragma unroll
    for (int e = 0; e < 8; ++e) {
      a[e] = al[e]; a[8 + e] = ah[e];
      b[e] = bl[e]; b[8 + e] = bh[e];
    }
    c = __builtin_amdgcn_wmma_f32_16x16x32_f16(false, a, false, b,
                                               (short)0, c, false, false);
  }
  {
    float bias = b1[wave * 16 + mr];
    #pragma unroll
    for (int r = 0; r < 8; ++r) {            // C: m = r + half*8, n = mr
      int m = r + half * 8;
      float v = c[r] + bias;
      v = v > 0.f ? v : 0.f;                 // relu
      h1s[m * H1_LD + wave * 16 + mr] = (_Float16)v;
    }
  }
  __syncthreads();

  // ---------------- stage 2 ----------------
  if (wave < 4) {
    const _Float16* w2row = w2h + (size_t)(wave * 16 + mr) * N_HID;
    v8f c2 = {};
    #pragma unroll
    for (int k0 = 0; k0 < N_HID; k0 += 32) {
      v16h a, b;
      h8 al = *(const h8*)(&h1s[mr * H1_LD + k0 + half * 8]);
      h8 ah = *(const h8*)(&h1s[mr * H1_LD + k0 + 16 + half * 8]);
      h8 bl = *(const h8*)(w2row + k0 + half * 8);
      h8 bh = *(const h8*)(w2row + k0 + 16 + half * 8);
      #pragma unroll
      for (int e = 0; e < 8; ++e) {
        a[e] = al[e]; a[8 + e] = ah[e];
        b[e] = bl[e]; b[8 + e] = bh[e];
      }
      c2 = __builtin_amdgcn_wmma_f32_16x16x32_f16(false, a, false, b,
                                                  (short)0, c2, false, false);
    }
    float bias = b2[wave * 16 + mr];
    #pragma unroll
    for (int r = 0; r < 8; ++r) {
      int m = r + half * 8;
      size_t idx = (m0 + m) * N_OUT + (wave * 16 + mr);
      float v = c2[r] + bias;
      h[idx] = v;      // teleport source
      z[idx] = v;      // z0
    }
  }
}

// ---------------------------------------------------------------------------
// Small helpers
// ---------------------------------------------------------------------------
__global__ void k_cvt_w(const float* __restrict__ W1, const float* __restrict__ W2,
                        _Float16* __restrict__ w1h, _Float16* __restrict__ w2h,
                        int n1, int n2)
{
  int t = blockIdx.x * blockDim.x + threadIdx.x;
  if (t < n1)            w1h[t]      = (_Float16)W1[t];
  else if (t < n1 + n2)  w2h[t - n1] = (_Float16)W2[t - n1];
}

__global__ void k_deg_init(float* __restrict__ deg, int n) {
  int i = blockIdx.x * blockDim.x + threadIdx.x;
  if (i < n) deg[i] = 1.0f;                       // self-loop
}

__global__ void k_deg_edge(const int* __restrict__ dst, float* __restrict__ deg, int e) {
  int i = blockIdx.x * blockDim.x + threadIdx.x;
  if (i < e) atomicAdd(&deg[dst[i]], 1.0f);
}

__global__ void k_rsqrt(float* __restrict__ d, int n) {
  int i = blockIdx.x * blockDim.x + threadIdx.x;
  if (i < n) d[i] = rsqrtf(fmaxf(d[i], 1e-12f));
}

// agg = dinv^2 * z   (self-loop term seeds the accumulator)
__global__ void k_agg_init(const f4* __restrict__ z, const float* __restrict__ dinv,
                           f4* __restrict__ agg, int total /* N*16 */) {
  int t = blockIdx.x * blockDim.x + threadIdx.x;
  if (t < total) {
    float s = dinv[t >> 4];
    agg[t] = (s * s) * z[t];
  }
}

// One wave per edge: agg[dst] += dinv[src]*dinv[dst] * z[src]  (64 f32 = 2/lane)
__global__ void k_edge(const int* __restrict__ src, const int* __restrict__ dst,
                       const float* __restrict__ dinv, const float* __restrict__ z,
                       float* __restrict__ agg, int e)
{
  int gid  = blockIdx.x * blockDim.x + threadIdx.x;
  int eid  = gid >> 5;
  int lane = gid & 31;
  if (eid >= e) return;
  int s = src[eid], d = dst[eid];
  float w = dinv[s] * dinv[d];
  f2 v = *(const f2*)(z + (size_t)s * N_OUT + lane * 2);
  float* p = agg + (size_t)d * N_OUT + lane * 2;
  atomicAdd(p,     w * v.x);
  atomicAdd(p + 1, w * v.y);
}

// z = (1-a)*agg + a*h ; then seed next round: agg = dinv^2 * z
__global__ void k_update(f4* __restrict__ agg, const f4* __restrict__ h,
                         const float* __restrict__ dinv, f4* __restrict__ z,
                         int total /* N*16 */) {
  int t = blockIdx.x * blockDim.x + threadIdx.x;
  if (t < total) {
    f4 zn = (1.0f - ALPHA) * agg[t] + ALPHA * h[t];
    z[t] = zn;
    float s = dinv[t >> 4];
    agg[t] = (s * s) * zn;
  }
}

// ---------------------------------------------------------------------------
static inline size_t align256(size_t x) { return (x + 255) & ~(size_t)255; }

extern "C" void kernel_launch(void* const* d_in, const int* in_sizes, int n_in,
                              void* d_out, int out_size, void* d_ws, size_t ws_size,
                              hipStream_t stream) {
  const float* x  = (const float*)d_in[0];
  const float* W1 = (const float*)d_in[1];
  const float* b1 = (const float*)d_in[2];
  const float* W2 = (const float*)d_in[3];
  const float* b2 = (const float*)d_in[4];
  const int*   ei = (const int*)d_in[5];
  float* z = (float*)d_out;

  const int N = in_sizes[0] / N_IN;     // 100000
  const int E = in_sizes[5] / 2;        // 1600000
  const int* src = ei;
  const int* dst = ei + E;

  // workspace carve
  char* p = (char*)d_ws;
  float* h    = (float*)p;  p += align256((size_t)N * N_OUT * sizeof(float));
  float* agg  = (float*)p;  p += align256((size_t)N * N_OUT * sizeof(float));
  float* dinv = (float*)p;  p += align256((size_t)N * sizeof(float));
  _Float16* w1h = (_Float16*)p; p += align256((size_t)N_HID * N_IN * sizeof(_Float16));
  _Float16* w2h = (_Float16*)p; p += align256((size_t)N_OUT * N_HID * sizeof(_Float16));

  const int n1 = N_HID * N_IN, n2 = N_OUT * N_HID;

  // 1) weights -> f16
  k_cvt_w<<<(n1 + n2 + 255) / 256, 256, 0, stream>>>(W1, W2, w1h, w2h, n1, n2);

  // 2) fused MLP (WMMA), writes h and z0
  k_mlp<<<N / 16, 512, 0, stream>>>(x, w1h, b1, w2h, b2, h, z);

  // 3) symmetric GCN norm: deg -> dinv (in place)
  k_deg_init<<<(N + 255) / 256, 256, 0, stream>>>(dinv, N);
  k_deg_edge<<<(E + 255) / 256, 256, 0, stream>>>(dst, dinv, E);
  k_rsqrt  <<<(N + 255) / 256, 256, 0, stream>>>(dinv, N);

  // 4) seed accumulator with self-loop term
  const int totv4 = N * (N_OUT / 4);
  k_agg_init<<<(totv4 + 255) / 256, 256, 0, stream>>>((const f4*)z, dinv, (f4*)agg, totv4);

  // 5) K = 5 propagation rounds
  const long long ethreads = (long long)E * 32;
  const int eblocks = (int)((ethreads + 255) / 256);
  for (int k = 0; k < 5; ++k) {
    k_edge  <<<eblocks, 256, 0, stream>>>(src, dst, dinv, z, agg, E);
    k_update<<<(totv4 + 255) / 256, 256, 0, stream>>>((f4*)agg, (const f4*)h,
                                                      dinv, (f4*)z, totv4);
  }
}
